// CFConvCluster_34093450396366
// MI455X (gfx1250) — matrix operand order repaired and checked by reading, not compiled, not run on hardware
//
#include <hip/hip_runtime.h>

#define DIM 64
#define RBF 128

typedef __attribute__((ext_vector_type(16))) __bf16 v16bf;
typedef __attribute__((ext_vector_type(8)))  float  v8f;

struct B32x8 { uint4 lo, hi; };   // 32 bytes == one v16bf fragment

// Pack two f32 into one dword of bf16 (RTE) with the CDNA5 packed convert.
// Single VALU op vs ~10 for the integer-emulated round-to-nearest-even.
__device__ __forceinline__ unsigned packbf(float lo, float hi) {
  unsigned r;
  asm("v_cvt_pk_bf16_f32 %0, %1, %2" : "=v"(r) : "v"(lo), "v"(hi));
  return r;
}
__device__ __forceinline__ unsigned short f2bf(float f) {
  return (unsigned short)(packbf(f, 0.0f) & 0xFFFFu);
}

// Load a pre-packed 8-dword fragment from LDS as v16bf (2x ds_load_b128)
__device__ __forceinline__ v16bf lds_frag(const unsigned* p) {
  B32x8 t;
  t.lo = *(const uint4*)(p);
  t.hi = *(const uint4*)(p + 4);
  return __builtin_bit_cast(v16bf, t);
}

// Build bf16 A fragment from two 8-float consecutive-k chunks (a,b = k..k+7; c,d = k+16..k+23)
__device__ __forceinline__ v16bf cvtA(const float4& a, const float4& b,
                                      const float4& c, const float4& d) {
  B32x8 t;
  t.lo.x = packbf(a.x, a.y); t.lo.y = packbf(a.z, a.w);
  t.lo.z = packbf(b.x, b.y); t.lo.w = packbf(b.z, b.w);
  t.hi.x = packbf(c.x, c.y); t.hi.y = packbf(c.z, c.w);
  t.hi.z = packbf(d.x, d.y); t.hi.w = packbf(d.z, d.w);
  return __builtin_bit_cast(v16bf, t);
}

// torch Softplus(beta=0.5, threshold=14): exact identity when 0.5*x > 14
__device__ __forceinline__ float softplus_sp(float x) {
  float bx = 0.5f * x;
  return (bx > 14.0f) ? x : 2.0f * log1pf(__expf(bx));
}

// ---------------------------------------------------------------------------
// out := 0
__global__ void zero_kernel(float* __restrict__ p, int n) {
  int i = blockIdx.x * 256 + threadIdx.x;
  if (i < n) p[i] = 0.0f;
}

__global__ void reset_kernel(unsigned* cnt) {
  if (threadIdx.x == 0) *cnt = 0u;
}

// Stream-compact masked edge ids into idx[], one atomic per wave (ballot)
__global__ void compact_kernel(const unsigned char* __restrict__ m,
                               unsigned* __restrict__ cnt,
                               int* __restrict__ idx, int n) {
  const int i = blockIdx.x * 256 + threadIdx.x;
  const bool act = (i < n) && (m[i] != 0);
  const unsigned long long bal = __ballot(act);
  if (bal == 0ull) return;
  const int lane = threadIdx.x & 31;
  const int leader = __ffsll((unsigned long long)bal) - 1;
  unsigned base = 0;
  if (lane == leader) base = atomicAdd(cnt, (unsigned)__popcll(bal));
  base = __shfl(base, leader, 32);
  if (act) {
    const unsigned pre = (unsigned)__popcll(bal & ((1ull << lane) - 1ull));
    idx[base + pre] = i;
  }
}

// ---------------------------------------------------------------------------
// Fused edge-MLP (bf16 WMMA, f32 accum) + masked gather-multiply-scatter.
// Block = 256 threads = 8 waves; each wave owns 16 edges x 4 tile-iterations
// => 512 edges per block.  Weights staged once per block into LDS in native
// CDNA5 B-fragment order.
template <bool COMPACT>
__global__ __launch_bounds__(256, 2) void cfconv_kernel(
    const float* __restrict__ rbf,
    const float* __restrict__ node,
    const int*   __restrict__ src,
    const int*   __restrict__ dst,
    const unsigned char* __restrict__ emask,
    const unsigned* __restrict__ cnt,
    const int*   __restrict__ idx,
    const float* __restrict__ W1,
    const float* __restrict__ b1,
    const float* __restrict__ W2,
    const float* __restrict__ b2,
    float* __restrict__ out,
    int n_edges)
{
  // B-fragment layout: [kstep][lane-half][n][j] ; dword j = packed rows
  // (32*kstep + 8*half + 2j + (j>=4 ? 8 : 0), +1) of column n.
  __shared__ __align__(16) unsigned        s_w1f[4][2][64][8];  // 16 KB
  __shared__ __align__(16) unsigned        s_w2f[2][2][64][8];  //  8 KB
  __shared__ __align__(16) unsigned short  s_h[8][16][64];      // 16 KB (per-wave h)

  const int tid   = threadIdx.x;
  const int work  = COMPACT ? (int)*cnt : n_edges;
  const int tile0 = (int)blockIdx.x * 512;
  if (tile0 >= work) return;

  #pragma unroll 4
  for (int i = tid; i < 4096; i += 256) {
    const int s = i >> 10, h = (i >> 9) & 1, n = (i >> 3) & 63, j = i & 7;
    const int k = 32*s + 8*h + 2*j + (j >= 4 ? 8 : 0);
    s_w1f[s][h][n][j] = packbf(W1[k*DIM + n], W1[(k+1)*DIM + n]);
  }
  #pragma unroll 4
  for (int i = tid; i < 2048; i += 256) {
    const int s = i >> 10, h = (i >> 9) & 1, n = (i >> 3) & 63, j = i & 7;
    const int k = 32*s + 8*h + 2*j + (j >= 4 ? 8 : 0);
    s_w2f[s][h][n][j] = packbf(W2[k*DIM + n], W2[(k+1)*DIM + n]);
  }
  __syncthreads();

  const int wave = tid >> 5;
  const int lane = tid & 31;
  const int lm   = lane & 15;   // A row (load) / C column
  const int hi   = lane >> 4;   // lane half -> K swizzle / C row group

  float b1v[4], b2v[4];
  #pragma unroll
  for (int t = 0; t < 4; ++t) { b1v[t] = b1[lm + 16*t]; b2v[t] = b2[lm + 16*t]; }

  const v8f zv = {0.f,0.f,0.f,0.f,0.f,0.f,0.f,0.f};

  for (int tt = 0; tt < 4; ++tt) {
    const int base = tile0 + (tt*8 + wave)*16;   // this wave's 16-edge tile
    if (base >= work) break;                     // wave-uniform

    // ---- A fragment source: rbf row of this lane's edge ----
    const int gA = base + lm;
    int eA;
    if constexpr (COMPACT) eA = (gA < work)    ? idx[gA] : 0;
    else                   eA = (gA < n_edges) ? gA      : 0;
    const float* arow = rbf + (size_t)eA * RBF + hi*8;

    if constexpr (!COMPACT) {                    // prefetch next tile's rows
      const int gP = base + 128 + lm;
      if (gP < n_edges) __builtin_prefetch(rbf + (size_t)gP * RBF + hi*8, 0, 0);
    }

    // ---- layer 1: [16x128]bf16 @ [128x64]bf16 -> f32, K in 4 steps of 32 ----
    v8f acc[4];
    #pragma unroll
    for (int t = 0; t < 4; ++t) acc[t] = zv;

    #pragma unroll
    for (int s = 0; s < 4; ++s) {
      const float4 f0 = *(const float4*)(arow + 32*s + 0);
      const float4 f1 = *(const float4*)(arow + 32*s + 4);
      const float4 f2 = *(const float4*)(arow + 32*s + 16);
      const float4 f3 = *(const float4*)(arow + 32*s + 20);
      const v16bf A = cvtA(f0, f1, f2, f3);
      #pragma unroll
      for (int t = 0; t < 4; ++t) {
        const v16bf B = lds_frag(&s_w1f[s][hi][lm + 16*t][0]);
        acc[t] = __builtin_amdgcn_wmma_f32_16x16x32_bf16(
                     false, A, false, B, (short)0, acc[t], false, false);
      }
    }

    // ---- bias + softplus; transpose C-layout -> A-layout via per-wave LDS ----
    #pragma unroll
    for (int t = 0; t < 4; ++t) {
      const int n = lm + 16*t;
      #pragma unroll
      for (int i = 0; i < 8; ++i)
        s_h[wave][hi*8 + i][n] = f2bf(softplus_sp(acc[t][i] + b1v[t]));
    }
    // same-wave DS store->load is in-order; no barrier needed

    // ---- layer 2: [16x64]bf16 @ [64x64]bf16 -> f32, K in 2 steps of 32 ----
    v8f acc2[4];
    #pragma unroll
    for (int t = 0; t < 4; ++t) acc2[t] = zv;

    #pragma unroll
    for (int s = 0; s < 2; ++s) {
      const unsigned short* hrow = &s_h[wave][lm][32*s + 8*hi];
      B32x8 tr;
      tr.lo = *(const uint4*)(hrow);        // k .. k+7
      tr.hi = *(const uint4*)(hrow + 16);   // k+16 .. k+23
      const v16bf A2 = __builtin_bit_cast(v16bf, tr);
      #pragma unroll
      for (int t = 0; t < 4; ++t) {
        const v16bf B = lds_frag(&s_w2f[s][hi][lm + 16*t][0]);
        acc2[t] = __builtin_amdgcn_wmma_f32_16x16x32_bf16(
                      false, A2, false, B, (short)0, acc2[t], false, false);
      }
    }

    // ---- bias + gather new_node[src]*h, scatter-add into out[dst] ----
    #pragma unroll
    for (int i = 0; i < 8; ++i) {
      const int g = base + hi*8 + i;        // edge of C row (m = 8*hi + i)
      int e; bool valid;
      if constexpr (COMPACT) { valid = (g < work); e = valid ? idx[g] : 0; }
      else { valid = (g < n_edges) && (emask[g] != 0); e = valid ? g : 0; }
      if (valid) {
        const int sb = src[e] * DIM;
        const int db = dst[e] * DIM;
        #pragma unroll
        for (int t = 0; t < 4; ++t) {
          const int n = lm + 16*t;
          const float v = (acc2[t][i] + b2v[t]) * node[sb + n];
          unsafeAtomicAdd(&out[db + n], v);   // global_atomic_add_f32 (L2-resident)
        }
      }
    }
  }
}

// ---------------------------------------------------------------------------
extern "C" void kernel_launch(void* const* d_in, const int* in_sizes, int n_in,
                              void* d_out, int out_size, void* d_ws, size_t ws_size,
                              hipStream_t stream)
{
  const float*         rbf   = (const float*)d_in[0];
  const float*         node  = (const float*)d_in[1];
  const int*           src   = (const int*)d_in[2];
  const int*           dst   = (const int*)d_in[3];
  const unsigned char* emask = (const unsigned char*)d_in[4];  // jnp bool = 1B
  const float*         W1    = (const float*)d_in[5];
  const float*         b1    = (const float*)d_in[6];
  const float*         W2    = (const float*)d_in[7];
  const float*         b2    = (const float*)d_in[8];
  float* out = (float*)d_out;
  const int E = in_sizes[2];

  unsigned* cnt = (unsigned*)d_ws;
  int*      idx = ((int*)d_ws) + 4;   // 16-byte offset past the counter
  const bool compact = (d_ws != nullptr) &&
                       (ws_size >= (size_t)(E + 8) * sizeof(int));

  zero_kernel<<<(out_size + 255) / 256, 256, 0, stream>>>(out, out_size);

  const int blocks = (E + 511) / 512;
  if (compact) {
    reset_kernel<<<1, 32, 0, stream>>>(cnt);
    compact_kernel<<<(E + 255) / 256, 256, 0, stream>>>(emask, cnt, idx, E);
    cfconv_kernel<true><<<blocks, 256, 0, stream>>>(
        rbf, node, src, dst, emask, cnt, idx, W1, b1, W2, b2, out, E);
  } else {
    cfconv_kernel<false><<<blocks, 256, 0, stream>>>(
        rbf, node, src, dst, emask, cnt, idx, W1, b1, W2, b2, out, E);
  }
}
